// Decoder_39359080301163
// MI455X (gfx1250) — matrix operand order repaired and checked by reading, not compiled, not run on hardware
//
#include <hip/hip_runtime.h>
#include <stdint.h>

// ---------------------------------------------------------------------------
// GRU decoder, fully fused, bf16 WMMA (gfx1250 / CDNA5, wave32)
// B=4096 T=64 V=512 E=32 H=128 FC=256 O=1024
// Each wave owns a 16-row batch tile; recurrence is per-row independent.
// ---------------------------------------------------------------------------

typedef __attribute__((ext_vector_type(16))) __bf16 v16bf;
typedef __attribute__((ext_vector_type(8)))  __bf16 v8bf;
typedef __attribute__((ext_vector_type(8)))  float  v8f;

#define B_   4096
#define T_   64
#define V_   512
#define E_   32
#define H_   128
#define FC_  256
#define O_   1024

// workspace (bf16) element offsets
#define WS_WEMB 0            // 32*512    = 16384
#define WS_WIH  16384        // 384*32    = 12288
#define WS_WHH  28672        // 384*128   = 49152
#define WS_WFC1 77824        // 256*128   = 32768
#define WS_WFC2 110592       // 1024*256  = 262144

#define WAVES    2
#define BLOCKSZ  (WAVES * 32)
#define GRID     (B_ / (WAVES * 16))          // 128 workgroups

// LDS layout (ushort elements)
#define LDS_WIH   0                           // 12288
#define LDS_WHH   12288                       // 49152
#define LDS_WVSZ  61440                       // shared weights total
#define PW_X      0                           // 16*32  = 512
#define PW_H0     512                         // 16*128 = 2048
#define PW_H1     2560                        // 16*128 = 2048
#define PW_HID    4608                        // 16*256 = 4096
#define PW_SZ     8704
#define LDS_USHORTS (LDS_WVSZ + WAVES * PW_SZ)
#define LDS_BYTES   (LDS_USHORTS * 2)         // 157,696 B

// native bf16 converts (v_cvt_pk_bf16_f32 path on gfx1250)
__device__ __forceinline__ unsigned short f2bf(float f) {
  union { __bf16 b; unsigned short u; } c;
  c.b = (__bf16)f;
  return c.u;
}
__device__ __forceinline__ float bf2f(unsigned short h) {
  union { __bf16 b; unsigned short u; } c;
  c.u = h;
  return (float)c.b;
}

// fast transcendentals: v_exp_f32 + v_rcp_f32 (co-executes with XDL WMMA)
__device__ __forceinline__ float sigm(float x) {
  return __builtin_amdgcn_rcpf(1.0f + __expf(-x));
}
__device__ __forceinline__ float tanh_fast(float x) {
  float xc = fminf(fmaxf(x, -15.0f), 15.0f);
  float t  = __expf(2.0f * xc);
  return (t - 1.0f) * __builtin_amdgcn_rcpf(t + 1.0f);
}

union ABu { v16bf v; v8bf h[2]; unsigned short u[16]; uint4 q[2]; };

// A operand: 16xK row-major bf16; lane L: M=L%16, g=L/16.
// halves[0..7]  = row[off + 8g .. +7], halves[8..15] = row[off + 16 + 8g .. +7]
__device__ __forceinline__ v16bf loadA_bf(const unsigned short* row, int off, int lg) {
  ABu r;
  r.q[0] = *(const uint4*)(row + off + lg * 8);
  r.q[1] = *(const uint4*)(row + off + 16 + lg * 8);
  return r.v;
}

// B operand: B[k][n] with W stored row-major [N][K]; lane L: N-col = n, g=L/16.
// halves[0..7] = W[n][off + 8g .. +7], halves[8..15] = W[n][off + 16 + 8g .. +7]
__device__ __forceinline__ v16bf loadB_bf(const unsigned short* W, int K, int n,
                                          int off, int lg) {
  const unsigned short* p = W + (size_t)n * K + off + lg * 8;
  ABu r;
  r.q[0] = *(const uint4*)(p);
  r.q[1] = *(const uint4*)(p + 16);
  return r.v;
}

// A operand from an fp32 row (message): non-temporal streaming load + packed cvt
__device__ __forceinline__ v16bf loadA_f32(const float* row, int off, int lg) {
  const v8f* p0 = (const v8f*)(row + off + lg * 8);
  const v8f* p1 = (const v8f*)(row + off + 16 + lg * 8);
  v8f f0 = __builtin_nontemporal_load(p0);
  v8f f1 = __builtin_nontemporal_load(p1);
  ABu r;
  r.h[0] = __builtin_convertvector(f0, v8bf);
  r.h[1] = __builtin_convertvector(f1, v8bf);
  return r.v;
}

__device__ __forceinline__ v8f wmma_bf16(v16bf a, v16bf b, v8f c) {
  return __builtin_amdgcn_wmma_f32_16x16x32_bf16(false, a, false, b, (short)0, c,
                                                 false, false);
}

__global__ void cvt_f32_bf16(const float* __restrict__ src,
                             unsigned short* __restrict__ dst, int n) {
  int i = blockIdx.x * 256 + threadIdx.x;
  if (i < n) dst[i] = f2bf(src[i]);
}

__global__ __launch_bounds__(BLOCKSZ)
void gru_decoder(const float* __restrict__ msg,
                 const float* __restrict__ b_emb,
                 const float* __restrict__ init_emb,
                 const float* __restrict__ b_ih,
                 const float* __restrict__ b_hh,
                 const float* __restrict__ b_fc1,
                 const float* __restrict__ b_fc2,
                 const unsigned short* __restrict__ wsbf,
                 float* __restrict__ out) {
  extern __shared__ unsigned short smem[];
  const int lane = threadIdx.x & 31;
  const int wave = threadIdx.x >> 5;
  const int lm = lane & 15;   // A: M row / B: N col / C: N col
  const int lg = lane >> 4;   // lane group

  // --- cooperatively stage W_ih + W_hh into LDS (contiguous in ws) ---
  {
    const uint4* src = (const uint4*)(wsbf + WS_WIH);
    uint4* dst = (uint4*)(smem);
    for (int i = threadIdx.x; i < (LDS_WVSZ / 8); i += BLOCKSZ) dst[i] = src[i];
  }
  __syncthreads();

  const unsigned short* sWih = smem + LDS_WIH;
  const unsigned short* sWhh = smem + LDS_WHH;
  unsigned short* wbase = smem + LDS_WVSZ + wave * PW_SZ;
  unsigned short* xls  = wbase + PW_X;
  unsigned short* hls0 = wbase + PW_H0;
  unsigned short* hls1 = wbase + PW_H1;
  unsigned short* hid  = wbase + PW_HID;

  const unsigned short* gWemb = wsbf + WS_WEMB;
  const unsigned short* gWfc1 = wsbf + WS_WFC1;
  const unsigned short* gWfc2 = wsbf + WS_WFC2;

  const int rowBase = blockIdx.x * (WAVES * 16) + wave * 16;

  // --- per-wave init: x <- init_emb broadcast (no b_emb, per reference); h0 <- 0
  for (int idx = lane; idx < 16 * E_; idx += 32) xls[idx] = f2bf(init_emb[idx & (E_ - 1)]);
  for (int idx = lane; idx < 16 * H_; idx += 32) hls0[idx] = 0;

  // --- preload per-lane biases (column = lm in every 16-wide N tile) ---
  float bR[8], bZ[8], bIN[8], bHN[8];
#pragma unroll
  for (int j = 0; j < 8; ++j) {
    int c = j * 16 + lm;
    bR[j]  = b_ih[c] + b_hh[c];
    bZ[j]  = b_ih[H_ + c] + b_hh[H_ + c];
    bIN[j] = b_ih[2 * H_ + c];
    bHN[j] = b_hh[2 * H_ + c];
  }
  const float bE0 = b_emb[lm];
  const float bE1 = b_emb[16 + lm];
  const float* mrow0 = msg + (size_t)(rowBase + lm) * (T_ * V_);

  // =================== GRU time loop (sequential, per-wave) ===================
#pragma unroll 1
  for (int t = 0; t < T_; ++t) {
    if (t > 0) {
      const float* mrow = mrow0 + (size_t)(t - 1) * V_;
      if (t < T_ - 1) {                                     // global_prefetch_b8
        __builtin_prefetch(mrow + V_, 0, 1);
        __builtin_prefetch(mrow + V_ + 256, 0, 1);
      }
      v8f a0, a1;
#pragma unroll
      for (int i = 0; i < 8; ++i) { a0[i] = bE0; a1[i] = bE1; }

      // depth-2 software pipeline: issue loads for kt+2 before WMMAs of kt
      v16bf aP[2], b0P[2], b1P[2];
#pragma unroll
      for (int s = 0; s < 2; ++s) {
        aP[s]  = loadA_f32(mrow, s * 32, lg);
        b0P[s] = loadB_bf(gWemb, V_, lm,      s * 32, lg);
        b1P[s] = loadB_bf(gWemb, V_, 16 + lm, s * 32, lg);
      }
#pragma unroll
      for (int kt = 0; kt < V_ / 32; ++kt) {
        const int cur = kt & 1;
        v16bf aT = aP[cur], b0T = b0P[cur], b1T = b1P[cur];
        if (kt < V_ / 32 - 2) {
          aP[cur]  = loadA_f32(mrow, (kt + 2) * 32, lg);
          b0P[cur] = loadB_bf(gWemb, V_, lm,      (kt + 2) * 32, lg);
          b1P[cur] = loadB_bf(gWemb, V_, 16 + lm, (kt + 2) * 32, lg);
        }
        a0 = wmma_bf16(aT, b0T, a0);
        a1 = wmma_bf16(aT, b1T, a1);
      }
#pragma unroll
      for (int i = 0; i < 8; ++i) {
        int m = i + 8 * lg;
        xls[m * E_ + lm]      = f2bf(a0[i]);
        xls[m * E_ + 16 + lm] = f2bf(a1[i]);
      }
    }

    const unsigned short* hcur = (t & 1) ? hls1 : hls0;
    unsigned short*       hnxt = (t & 1) ? hls0 : hls1;

    v16bf xA  = loadA_bf(xls  + lm * E_, 0,  lg);
    v16bf hA0 = loadA_bf(hcur + lm * H_, 0,  lg);
    v16bf hA1 = loadA_bf(hcur + lm * H_, 32, lg);
    v16bf hA2 = loadA_bf(hcur + lm * H_, 64, lg);
    v16bf hA3 = loadA_bf(hcur + lm * H_, 96, lg);

#pragma unroll
    for (int j = 0; j < 8; ++j) {
      const int nR = j * 16 + lm;
      const int nZ = H_ + nR;
      const int nN = 2 * H_ + nR;
      v8f aR, aZ, aI, aH;
#pragma unroll
      for (int i = 0; i < 8; ++i) { aR[i] = bR[j]; aZ[i] = bZ[j]; aI[i] = bIN[j]; aH[i] = bHN[j]; }
      // input-side gates (K = E = 32 -> single WMMA each)
      aR = wmma_bf16(xA, loadB_bf(sWih, E_, nR, 0, lg), aR);
      aZ = wmma_bf16(xA, loadB_bf(sWih, E_, nZ, 0, lg), aZ);
      aI = wmma_bf16(xA, loadB_bf(sWih, E_, nN, 0, lg), aI);
      // hidden-side gates (K = H = 128 -> 4 WMMAs each)
#pragma unroll
      for (int kt = 0; kt < 4; ++kt) {
        v16bf hA = (kt == 0) ? hA0 : (kt == 1) ? hA1 : (kt == 2) ? hA2 : hA3;
        aR = wmma_bf16(hA, loadB_bf(sWhh, H_, nR, kt * 32, lg), aR);
        aZ = wmma_bf16(hA, loadB_bf(sWhh, H_, nZ, kt * 32, lg), aZ);
        aH = wmma_bf16(hA, loadB_bf(sWhh, H_, nN, kt * 32, lg), aH);
      }
      // elementwise GRU update on C-layout tiles
#pragma unroll
      for (int i = 0; i < 8; ++i) {
        int m = i + 8 * lg;
        float r  = sigm(aR[i]);
        float z  = sigm(aZ[i]);
        float nn = tanh_fast(aI[i] + r * aH[i]);
        float ho = bf2f(hcur[m * H_ + j * 16 + lm]);
        hnxt[m * H_ + j * 16 + lm] = f2bf((1.0f - z) * nn + z * ho);
      }
    }
  }

  // =================== fc1 (ELU) -> hid ===================
  const unsigned short* hf = hls0;  // final h (t=63 wrote buffer 0)
  {
    v16bf hA0 = loadA_bf(hf + lm * H_, 0,  lg);
    v16bf hA1 = loadA_bf(hf + lm * H_, 32, lg);
    v16bf hA2 = loadA_bf(hf + lm * H_, 64, lg);
    v16bf hA3 = loadA_bf(hf + lm * H_, 96, lg);
#pragma unroll
    for (int nt = 0; nt < FC_ / 16; ++nt) {
      float bias = b_fc1[nt * 16 + lm];
      v8f acc;
#pragma unroll
      for (int i = 0; i < 8; ++i) acc[i] = bias;
      acc = wmma_bf16(hA0, loadB_bf(gWfc1, H_, nt * 16 + lm, 0,  lg), acc);
      acc = wmma_bf16(hA1, loadB_bf(gWfc1, H_, nt * 16 + lm, 32, lg), acc);
      acc = wmma_bf16(hA2, loadB_bf(gWfc1, H_, nt * 16 + lm, 64, lg), acc);
      acc = wmma_bf16(hA3, loadB_bf(gWfc1, H_, nt * 16 + lm, 96, lg), acc);
#pragma unroll
      for (int i = 0; i < 8; ++i) {
        float v = acc[i];
        float e = (v > 0.0f) ? v : (__expf(v) - 1.0f);
        hid[(i + 8 * lg) * FC_ + nt * 16 + lm] = f2bf(e);
      }
    }
  }

  // =================== fc2 (sigmoid) -> out ===================
  {
    v16bf dA[8];
#pragma unroll
    for (int kt = 0; kt < 8; ++kt) dA[kt] = loadA_bf(hid + lm * FC_, kt * 32, lg);
#pragma unroll 2
    for (int nt = 0; nt < O_ / 16; ++nt) {
      float bias = b_fc2[nt * 16 + lm];
      v8f acc;
#pragma unroll
      for (int i = 0; i < 8; ++i) acc[i] = bias;
#pragma unroll
      for (int kt = 0; kt < FC_ / 32; ++kt)
        acc = wmma_bf16(dA[kt], loadB_bf(gWfc2, FC_, nt * 16 + lm, kt * 32, lg), acc);
#pragma unroll
      for (int i = 0; i < 8; ++i) {
        int m = i + 8 * lg;
        __builtin_nontemporal_store(sigm(acc[i]),
            out + (size_t)(rowBase + m) * O_ + nt * 16 + lm);
      }
    }
  }
}

// ---------------------------------------------------------------------------

extern "C" void kernel_launch(void* const* d_in, const int* in_sizes, int n_in,
                              void* d_out, int out_size, void* d_ws, size_t ws_size,
                              hipStream_t stream) {
  (void)in_sizes; (void)n_in; (void)out_size; (void)ws_size;
  const float* msg      = (const float*)d_in[0];
  const float* W_emb    = (const float*)d_in[1];
  const float* b_emb    = (const float*)d_in[2];
  const float* init_emb = (const float*)d_in[3];
  const float* W_ih     = (const float*)d_in[4];
  const float* W_hh     = (const float*)d_in[5];
  const float* b_ih     = (const float*)d_in[6];
  const float* b_hh     = (const float*)d_in[7];
  const float* W_fc1    = (const float*)d_in[8];
  const float* b_fc1    = (const float*)d_in[9];
  const float* W_fc2    = (const float*)d_in[10];
  const float* b_fc2    = (const float*)d_in[11];
  float* out = (float*)d_out;
  unsigned short* wsbf = (unsigned short*)d_ws;

  // fp32 -> bf16 weight conversion into workspace (L2-resident thereafter)
  auto cvt = [&](const float* src, int off, int n) {
    cvt_f32_bf16<<<(n + 255) / 256, 256, 0, stream>>>(src, wsbf + off, n);
  };
  cvt(W_emb, WS_WEMB, E_ * V_);
  cvt(W_ih,  WS_WIH,  3 * H_ * E_);
  cvt(W_hh,  WS_WHH,  3 * H_ * H_);
  cvt(W_fc1, WS_WFC1, FC_ * H_);
  cvt(W_fc2, WS_WFC2, O_ * FC_);

  gru_decoder<<<GRID, BLOCKSZ, LDS_BYTES, stream>>>(
      msg, b_emb, init_emb, b_ih, b_hh, b_fc1, b_fc2, wsbf, out);
}
